// ConvAttention_18949395710181
// MI455X (gfx1250) — matrix operand order repaired and checked by reading, not compiled, not run on hardware
//
#include <hip/hip_runtime.h>
#include <hip/hip_bf16.h>

// ---------------------------------------------------------------------------
// ConvAttention energy block for MI455X (gfx1250, wave32, WMMA bf16 16x16x32)
//
// ~27 GFLOP, intermediates fit in L2 -> matrix-core bound. All four GEMMs
// (qkv conv, Q.K^T, P.V, out conv) run on v_wmma_f32_16x16x32_bf16 with fp32
// accumulation; softmax / logsumexp / energies in fp32.
//
// Fragment I/O: all LDS operand tiles are stored K-contiguous so both A and B
// fragments load as exactly 2x ds_load_b128 per lane (ISA 7.12.2: both lane
// layouts are contiguous K runs).  V staging uses GLOBAL_LOAD_ASYNC_TO_LDS
// when the toolchain exposes the builtin (ASYNCcnt path), sync b128 otherwise.
// ---------------------------------------------------------------------------

#define BATCH  8
#define CIN    256
#define SIDE   48
#define HW     (SIDE * SIDE)        // 2304
#define HEADS  4
#define DHEAD  32
#define HID    (HEADS * DHEAD)      // 128
#define QKVC   (3 * HID)            // 384
#define ATTN_SCALE 0.17677669529663689f   // 32^-0.5

#define S32  40    // padded stride (elements) for 32-wide K tiles (16B mult)
#define S128 136   // padded stride (elements) for 128-wide K tiles (16B mult)

#if defined(__has_builtin)
#  if __has_builtin(__builtin_amdgcn_global_load_async_to_lds_b128)
#    define HAVE_ASYNC_LDS 1
#  endif
#endif
#ifndef HAVE_ASYNC_LDS
#  define HAVE_ASYNC_LDS 0
#endif

typedef __attribute__((ext_vector_type(16))) __bf16          v16bf;
typedef __attribute__((ext_vector_type(16))) unsigned short  v16us;
typedef __attribute__((ext_vector_type(8)))  unsigned short  v8us;
typedef __attribute__((ext_vector_type(8)))  float           v8f;

#if HAVE_ASYNC_LDS
// Exact parameter types per hipcc diagnostic: pointer to 16B int vector,
// global (AS1) source / LDS (AS3) destination.
typedef __attribute__((vector_size(4 * sizeof(int)))) int        async_v4i;
typedef async_v4i __attribute__((address_space(1)))*             async_gp;
typedef async_v4i __attribute__((address_space(3)))*             async_lp;
#endif

// ----- bf16 <-> f32 (round to nearest even), storage as ushort bits --------
__device__ __forceinline__ unsigned short f2bf(float f) {
  unsigned int x = __float_as_uint(f);
  unsigned int r = x + 0x7fffu + ((x >> 16) & 1u);
  return (unsigned short)(r >> 16);
}

__device__ __forceinline__ v8f vzero() {
  v8f z;
#pragma unroll
  for (int i = 0; i < 8; ++i) z[i] = 0.0f;
  return z;
}

__device__ __forceinline__ v8f wmma_bf16(v16bf a, v16bf b, v8f c) {
  return __builtin_amdgcn_wmma_f32_16x16x32_bf16(false, a, false, b,
                                                 (short)0, c, false, false);
}

__device__ __forceinline__ v16bf combine(v8us lo, v8us hi) {
  v16us u;
#pragma unroll
  for (int i = 0; i < 8; ++i) { u[i] = lo[i]; u[8 + i] = hi[i]; }
  return __builtin_bit_cast(v16bf, u);
}

// A fragment (16x32 MxK) from row-major K-contiguous LDS tile, stride S elems
// (S multiple of 8): lane m = L&15, kb = 8*(L>>4); elems 0-7 = K kb..kb+7,
// elems 8-15 = K kb+16..kb+23  ->  2x ds_load_b128.
__device__ __forceinline__ v16bf frag_a_row(const unsigned short* base, int S) {
  int L = threadIdx.x & 31;
  const unsigned short* p = base + (L & 15) * S + ((L >> 4) << 3);
  v8us lo = *(const v8us*)p;
  v8us hi = *(const v8us*)(p + 16);
  return combine(lo, hi);
}

// B fragment (32x16 KxN) from N-major K-contiguous LDS tile, stride S elems
// (S multiple of 8): lane n = L&15, kb = 8*(L>>4); elems 0-7 = K kb..kb+7,
// elems 8-15 = K kb+16..kb+23  ->  2x ds_load_b128.
__device__ __forceinline__ v16bf frag_b_nmaj(const unsigned short* base, int S) {
  int L = threadIdx.x & 31;
  const unsigned short* p = base + (L & 15) * S + ((L >> 4) << 3);
  v8us lo = *(const v8us*)p;
  v8us hi = *(const v8us*)(p + 16);
  return combine(lo, hi);
}

// ---------------------------------------------------------------------------
__global__ void ConvAttn_zero_kernel(float* e) {
  if (threadIdx.x == 0) *e = 0.0f;
}

// ---------------------------------------------------------------------------
// K1: qkv = W_qkv @ x + b ; e_qkv ; emit bf16 q(*scale)/k/v  [b][128][2304]
// grid: (Ntile=9, Mtile=24, batch=8), block 256 (8 waves, 16x32 tile each:
// one A fragment reused across two B fragments -> 2 wmma / k-chunk / wave).
// ---------------------------------------------------------------------------
__global__ __launch_bounds__(256)
void ConvAttn_qkv_kernel(const float* __restrict__ x,
                         const float* __restrict__ w,
                         const float* __restrict__ bias,
                         unsigned short* __restrict__ qb,
                         unsigned short* __restrict__ kb,
                         unsigned short* __restrict__ vb,
                         float* __restrict__ e_acc) {
  __shared__ __align__(16) unsigned short sW[16 * S32];    // [m][k]
  __shared__ __align__(16) unsigned short sX[256 * S32];   // [n][k] (transposed)
  const int b = blockIdx.z, mt = blockIdx.y, nt = blockIdx.x;
  const int tid = threadIdx.x, wave = tid >> 5, lane = tid & 31;
  const int ntb = nt * 256;
  const float* xb = x + (size_t)b * CIN * HW + ntb;

  v8f acc0 = vzero(), acc1 = vzero();
  for (int kc = 0; kc < CIN; kc += 32) {
    for (int i = tid; i < 16 * 32; i += 256) {
      int m = i >> 5, k = i & 31;
      sW[m * S32 + k] = f2bf(w[(size_t)(mt * 16 + m) * CIN + kc + k]);
    }
    // X tile transpose-stage: coalesced float4 global reads, scatter b16 LDS
    for (int i = tid; i < 32 * 256 / 4; i += 256) {
      int k = i >> 6, n4 = (i & 63) * 4;
      const float4 f = *(const float4*)(xb + (size_t)(kc + k) * HW + n4);
      sX[(n4 + 0) * S32 + k] = f2bf(f.x);
      sX[(n4 + 1) * S32 + k] = f2bf(f.y);
      sX[(n4 + 2) * S32 + k] = f2bf(f.z);
      sX[(n4 + 3) * S32 + k] = f2bf(f.w);
    }
    __syncthreads();
    v16bf af = frag_a_row(sW, S32);
    acc0 = wmma_bf16(af, frag_b_nmaj(sX + (wave * 32)      * S32, S32), acc0);
    acc1 = wmma_bf16(af, frag_b_nmaj(sX + (wave * 32 + 16) * S32, S32), acc1);
    __syncthreads();
  }

  // q/k/v segment is uniform per block (mt tiles never straddle a boundary)
  unsigned short* dst = (mt < 8) ? qb : (mt < 16) ? kb : vb;
  const float mul = (mt < 8) ? ATTN_SCALE : 1.0f;
  const int cbase = (mt & 7) * 16;
  const int half = lane >> 4, nloc = lane & 15;
  const int n0 = ntb + wave * 32 + nloc, n1 = n0 + 16;
  const size_t obase = (size_t)b * HID * HW;
  float esum = 0.0f;
#pragma unroll
  for (int v = 0; v < 8; ++v) {
    int m = mt * 16 + v + 8 * half;               // output channel [0,384)
    int c = cbase + v + 8 * half;                 // channel within segment
    float y0 = acc0[v] + bias[m];
    float y1 = acc1[v] + bias[m];
    esum += -0.5f * (y0 * y0 + y1 * y1);          // energy on unscaled qkv
    dst[obase + (size_t)c * HW + n0] = f2bf(y0 * mul);
    dst[obase + (size_t)c * HW + n1] = f2bf(y1 * mul);
  }
#pragma unroll
  for (int mask = 16; mask; mask >>= 1) esum += __shfl_xor(esum, mask, 32);
  if (lane == 0) atomicAdd(e_acc, esum);
}

// ---------------------------------------------------------------------------
// K2: flash-style attention per (b,h); 4 waves x 16 rows; 128-col K/V chunks.
// Tracks m (max), l (sum exp), t (sum (s-m)e^{s-m}) per row for the entropy
// energy term  e_attn_row = t/l - log l.
// grid: (rowblocks=36, bh=32), block 128.
// ---------------------------------------------------------------------------
__global__ __launch_bounds__(128)
void ConvAttn_attn_kernel(const unsigned short* __restrict__ qb,
                          const unsigned short* __restrict__ kb,
                          const unsigned short* __restrict__ vb,
                          unsigned short* __restrict__ oc,
                          float* __restrict__ e_acc) {
  __shared__ __align__(16) unsigned short sQ[4 * 16 * S32];   // per-wave [m][d]
  __shared__ __align__(16) unsigned short sKT[128 * S32];     // [j][d]
  __shared__ __align__(16) unsigned short sV[32 * S128];      // [d][j]
  __shared__ __align__(16) unsigned short sP[4][16 * S128];   // per-wave [m][j]

  const int bh = blockIdx.y, b = bh >> 2, h = bh & 3;
  const int tid = threadIdx.x, wave = tid >> 5, lane = tid & 31;
  const int half = lane >> 4, nloc = lane & 15;
  const size_t chw = (size_t)HID * HW;
  const unsigned short* qbase = qb + (size_t)b * chw + (size_t)(h * DHEAD) * HW;
  const unsigned short* kbase = kb + (size_t)b * chw + (size_t)(h * DHEAD) * HW;
  const unsigned short* vbase = vb + (size_t)b * chw + (size_t)(h * DHEAD) * HW;
  const int rowbase0 = blockIdx.x * 64;
  const int rowbase  = rowbase0 + wave * 16;

  // stage Q (64 rows x 32 d) transposed: coalesced b128 reads over row index
  for (int i = tid; i < 64 * 32 / 8; i += 128) {
    int d = i >> 3, m8 = (i & 7) * 8;
    v8us qv = *(const v8us*)(qbase + (size_t)d * HW + rowbase0 + m8);
#pragma unroll
    for (int e = 0; e < 8; ++e) {
      int m = m8 + e;
      sQ[(m >> 4) * 16 * S32 + (m & 15) * S32 + d] = qv[e];
    }
  }
  __syncthreads();
  v16bf qf = frag_a_row(sQ + wave * 16 * S32, S32);   // A[m][k=d]

  float mrow[8], lrow[8], trow[8];
#pragma unroll
  for (int v = 0; v < 8; ++v) { mrow[v] = -1.0e30f; lrow[v] = 0.f; trow[v] = 0.f; }
  v8f o0 = vzero(), o1 = vzero();

  for (int jc = 0; jc < HW; jc += 128) {
    // stage K (transposed [j][d], manual) and V ([d][j], async DMA when avail)
    for (int i = tid; i < 32 * 128 / 8; i += 128) {
      int d = i >> 4, j8 = (i & 15) * 8;
      v8us kv = *(const v8us*)(kbase + (size_t)d * HW + jc + j8);
#pragma unroll
      for (int e = 0; e < 8; ++e) sKT[(j8 + e) * S32 + d] = kv[e];
#if HAVE_ASYNC_LDS
      __builtin_amdgcn_global_load_async_to_lds_b128(
          (async_gp)(vbase + (size_t)d * HW + jc + j8),
          (async_lp)(sV + d * S128 + j8),
          0, 0);
#else
      v8us vv = *(const v8us*)(vbase + (size_t)d * HW + jc + j8);
      *(v8us*)(sV + d * S128 + j8) = vv;
#endif
    }
#if HAVE_ASYNC_LDS
    asm volatile("s_wait_asynccnt 0" ::: "memory");
#endif
    __syncthreads();

    v8f s[8];
#pragma unroll
    for (int t = 0; t < 8; ++t)
      s[t] = wmma_bf16(qf, frag_b_nmaj(sKT + t * 16 * S32, S32), vzero());

    // per-row chunk max -> reduce across the 16 lanes of each half-wave
    float cmax[8];
#pragma unroll
    for (int v = 0; v < 8; ++v) {
      float mx = s[0][v];
#pragma unroll
      for (int t = 1; t < 8; ++t) mx = fmaxf(mx, s[t][v]);
      cmax[v] = mx;
    }
#pragma unroll
    for (int mask = 1; mask < 16; mask <<= 1)
#pragma unroll
      for (int v = 0; v < 8; ++v)
        cmax[v] = fmaxf(cmax[v], __shfl_xor(cmax[v], mask, 32));

#pragma unroll
    for (int v = 0; v < 8; ++v) {
      float mn = fmaxf(mrow[v], cmax[v]);
      float dm = mrow[v] - mn;
      float sc = expf(dm);
      trow[v] = sc * (trow[v] + dm * lrow[v]);   // shift-correct entropy accum
      lrow[v] *= sc;
      mrow[v] = mn;
      o0[v] *= sc; o1[v] *= sc;
    }

    // P = exp(s - m); accumulate l,t; stage P (bf16) to per-wave LDS
    float ladd[8], tadd[8];
#pragma unroll
    for (int v = 0; v < 8; ++v) { ladd[v] = 0.f; tadd[v] = 0.f; }
#pragma unroll
    for (int t = 0; t < 8; ++t) {
#pragma unroll
      for (int v = 0; v < 8; ++v) {
        float sv = s[t][v] - mrow[v];
        float p  = expf(sv);
        ladd[v] += p;
        tadd[v] += sv * p;
        sP[wave][(v + 8 * half) * S128 + t * 16 + nloc] = f2bf(p);
      }
    }
#pragma unroll
    for (int mask = 1; mask < 16; mask <<= 1)
#pragma unroll
      for (int v = 0; v < 8; ++v) {
        ladd[v] += __shfl_xor(ladd[v], mask, 32);
        tadd[v] += __shfl_xor(tadd[v], mask, 32);
      }
#pragma unroll
    for (int v = 0; v < 8; ++v) { lrow[v] += ladd[v]; trow[v] += tadd[v]; }

    // same-wave LDS store -> fragment reload hazard
    asm volatile("s_wait_dscnt 0" ::: "memory");

    // O += P @ V : A = P[16x32 subchunk], B = V^T halves (d 0-15 / 16-31)
#pragma unroll
    for (int kk = 0; kk < 4; ++kk) {
      v16bf pf  = frag_a_row(&sP[wave][kk * 32], S128);
      v16bf vf0 = frag_b_nmaj(sV + kk * 32, S128);
      v16bf vf1 = frag_b_nmaj(sV + 16 * S128 + kk * 32, S128);
      o0 = wmma_bf16(pf, vf0, o0);
      o1 = wmma_bf16(pf, vf1, o1);
    }
    __syncthreads();
  }

  // epilogue: normalize, write O as [b][h*32+d][i] bf16 for the output conv
#pragma unroll
  for (int v = 0; v < 8; ++v) {
    float inv = 1.0f / lrow[v];
    int i = rowbase + v + 8 * half;
    oc[(size_t)b * chw + (size_t)(h * DHEAD + nloc)      * HW + i] = f2bf(o0[v] * inv);
    oc[(size_t)b * chw + (size_t)(h * DHEAD + 16 + nloc) * HW + i] = f2bf(o1[v] * inv);
  }
  float e = 0.0f;
  if (nloc == 0) {
#pragma unroll
    for (int v = 0; v < 8; ++v) e += trow[v] / lrow[v] - logf(lrow[v]);
  }
#pragma unroll
  for (int mask = 16; mask; mask >>= 1) e += __shfl_xor(e, mask, 32);
  if (lane == 0) atomicAdd(e_acc, e);
}

// ---------------------------------------------------------------------------
// K3: out = W_out @ O + b_out ; e_out ; fp32 result straight to d_out
// grid: (Ntile=9, Mtile=16, batch=8), block 256 (16x32 tile per wave).
// ---------------------------------------------------------------------------
__global__ __launch_bounds__(256)
void ConvAttn_out_kernel(const unsigned short* __restrict__ oc,
                         const float* __restrict__ w,
                         const float* __restrict__ bias,
                         float* __restrict__ out,
                         float* __restrict__ e_acc) {
  __shared__ __align__(16) unsigned short sW[16 * S32];    // [m][k]
  __shared__ __align__(16) unsigned short sX[256 * S32];   // [n][k]
  const int b = blockIdx.z, mt = blockIdx.y, nt = blockIdx.x;
  const int tid = threadIdx.x, wave = tid >> 5, lane = tid & 31;
  const int ntb = nt * 256;
  const unsigned short* ob = oc + (size_t)b * HID * HW + ntb;

  v8f acc0 = vzero(), acc1 = vzero();
  for (int kc = 0; kc < HID; kc += 32) {
    for (int i = tid; i < 16 * 32; i += 256) {
      int m = i >> 5, k = i & 31;
      sW[m * S32 + k] = f2bf(w[(size_t)(mt * 16 + m) * HID + kc + k]);
    }
    for (int i = tid; i < 32 * 256 / 8; i += 256) {
      int k = i >> 5, n8 = (i & 31) * 8;
      v8us d8 = *(const v8us*)(ob + (size_t)(kc + k) * HW + n8);
#pragma unroll
      for (int e = 0; e < 8; ++e) sX[(n8 + e) * S32 + k] = d8[e];
    }
    __syncthreads();
    v16bf af = frag_a_row(sW, S32);
    acc0 = wmma_bf16(af, frag_b_nmaj(sX + (wave * 32)      * S32, S32), acc0);
    acc1 = wmma_bf16(af, frag_b_nmaj(sX + (wave * 32 + 16) * S32, S32), acc1);
    __syncthreads();
  }

  const int half = lane >> 4, nloc = lane & 15;
  const int n0 = ntb + wave * 32 + nloc, n1 = n0 + 16;
  float esum = 0.0f;
#pragma unroll
  for (int v = 0; v < 8; ++v) {
    int m = mt * 16 + v + 8 * half;
    float y0 = acc0[v] + bias[m];
    float y1 = acc1[v] + bias[m];
    esum += -0.5f * (y0 * y0 + y1 * y1);
    out[((size_t)b * CIN + m) * HW + n0] = y0;
    out[((size_t)b * CIN + m) * HW + n1] = y1;
  }
#pragma unroll
  for (int mask = 16; mask; mask >>= 1) esum += __shfl_xor(esum, mask, 32);
  if (lane == 0) atomicAdd(e_acc, esum);
}

// ---------------------------------------------------------------------------
__global__ void ConvAttn_fin_kernel(float* out, const float* e) {
  if (threadIdx.x == 0) out[(size_t)BATCH * CIN * HW] = *e;
}

// ---------------------------------------------------------------------------
extern "C" void kernel_launch(void* const* d_in, const int* in_sizes, int n_in,
                              void* d_out, int out_size, void* d_ws, size_t ws_size,
                              hipStream_t stream) {
  (void)in_sizes; (void)n_in; (void)out_size; (void)ws_size;
  const float* x     = (const float*)d_in[0];
  const float* w_qkv = (const float*)d_in[1];
  const float* b_qkv = (const float*)d_in[2];
  const float* w_out = (const float*)d_in[3];
  const float* b_out = (const float*)d_in[4];
  float* out = (float*)d_out;

  // workspace layout: [energy f32 | q bf16 | k bf16 | v bf16 | O bf16]
  const size_t QELEMS = (size_t)BATCH * HID * HW;      // 2,359,296
  char* wsp = (char*)d_ws;
  float* e_acc        = (float*)wsp;
  unsigned short* qbp = (unsigned short*)(wsp + 256);
  unsigned short* kbp = qbp + QELEMS;
  unsigned short* vbp = kbp + QELEMS;
  unsigned short* obp = vbp + QELEMS;                  // total ~18.9 MB

  ConvAttn_zero_kernel<<<1, 32, 0, stream>>>(e_acc);
  ConvAttn_qkv_kernel<<<dim3(HW / 256, QKVC / 16, BATCH), 256, 0, stream>>>(
      x, w_qkv, b_qkv, qbp, kbp, vbp, e_acc);
  ConvAttn_attn_kernel<<<dim3(HW / 64, BATCH * HEADS, 1), 128, 0, stream>>>(
      qbp, kbp, vbp, obp, e_acc);
  ConvAttn_out_kernel<<<dim3(HW / 256, CIN / 16, BATCH), 256, 0, stream>>>(
      obp, w_out, b_out, out, e_acc);
  ConvAttn_fin_kernel<<<1, 32, 0, stream>>>(out, e_acc);
}
